// AgnosticRoIExtractor_13924283974113
// MI455X (gfx1250) — compile-verified, matrix operand order checked by reading, not run on hardware
//
#include <hip/hip_runtime.h>
#include <hip/hip_bf16.h>
#include <math.h>

typedef __attribute__((ext_vector_type(16))) _Float16 v16h;
typedef __attribute__((ext_vector_type(8)))  float    v8f;

#define BB      16
#define NN      8192
#define CCLS    91
#define NCLS    90
#define TOTC    (NN * NCLS)          // 737280 candidates / image
#define PRE     1000
#define DETK    100
#define CAP     4096
#define NBINS   4096
#define IMGW    1216.0f
#define IMGH    800.0f
#define CLIPV   4.135166556742356f   // log(1000/16)
#define SCORE_T 0.05f
#define NMS_T   0.5f
#define MINSZ   0.01f
#define CSTRIDE 1024                 // padded candidate stride

// ---------------------------------------------------------------------------
// K1: per-proposal softmax statistics. Row max via lane halves + shfl_xor,
// denominator = row-sum of exp() computed on the matrix pipe:
// D = A(16x32 f16 exp chunk) * B(ones) accumulated f32 over 3 K-chunks.
// Wave handles 16 proposals: lane L -> row m = L&15, half = L>>4.
// Stores 1/denom so the 11.8M-candidate kernel multiplies instead of divides.
// Tail masking is branchless (clamped index + select) so no EXEC juggling.
// ---------------------------------------------------------------------------
__global__ __launch_bounds__(256) void k1_softmax_stats(
    const float* __restrict__ logits,
    float* __restrict__ rowmax_g,
    float* __restrict__ inv_denom_g) {
  __shared__ float sum_lds[8][16];
  const int wave = threadIdx.x >> 5;
  const int lane = threadIdx.x & 31;
  const int m    = lane & 15;
  const int half = lane >> 4;
  const int row  = blockIdx.x * 128 + wave * 16 + m;   // flat proposal (img*N+p)
  const float* lrow = logits + (long)row * CCLS;

  // row max (half 0 takes even classes, half 1 odd; combine across halves)
  float mx = -3.0e38f;
  for (int c = half; c < CCLS; c += 2) mx = fmaxf(mx, lrow[c]);
  mx = fmaxf(mx, __shfl_xor(mx, 16, 32));

  // ones B-fragment (layout irrelevant: all elements equal)
  v16h onesb;
#pragma unroll
  for (int e = 0; e < 16; ++e) onesb[e] = (_Float16)1.0f;

  v8f acc = {};
#pragma unroll
  for (int kb = 0; kb < 96; kb += 32) {
    v16h a;
#pragma unroll
    for (int e = 0; e < 16; ++e) {
      // ISA 16-bit A 16x32 layout: e<8 -> K = half*8+e ; e>=8 -> K = 16+half*8+(e-8)
      int k = kb + half * 8 + e + ((e < 8) ? 0 : 8);
      int kc = (k < CCLS) ? k : 0;          // clamped: load always in-bounds
      float v = __expf(lrow[kc] - mx);      // straight-line, no branch
      v = (k < CCLS) ? v : 0.0f;            // v_cndmask, not exec mask
      a[e] = (_Float16)v;
    }
    acc = __builtin_amdgcn_wmma_f32_16x16x32_f16(
        false, a, false, onesb, (short)0, acc, false, false);
  }

  // D layout: lanes 0-15 VGPR r = row r ; lanes 16-31 VGPR r = row r+8
  if (lane == 0) {
#pragma unroll
    for (int r = 0; r < 8; ++r) sum_lds[wave][r] = acc[r];
  }
  if (lane == 16) {
#pragma unroll
    for (int r = 0; r < 8; ++r) sum_lds[wave][8 + r] = acc[r];
  }
  __syncthreads();
  if (half == 0) {
    rowmax_g[row]    = mx;
    inv_denom_g[row] = 1.0f / sum_lds[wave][m];
  }
}

// ---------------------------------------------------------------------------
// K2: per-candidate score + decode + mask, write masked score, histogram.
// Grid: (TOTC/256, BB) -> no 64-bit div/mod for image id.
// ---------------------------------------------------------------------------
__device__ __forceinline__ void decode_clip(const float4 r4, const float4 pr,
                                            float& x1, float& y1, float& x2, float& y2) {
  float w  = pr.z - pr.x, h = pr.w - pr.y;
  float cx = pr.x + 0.5f * w, cy = pr.y + 0.5f * h;
  float dx = r4.x * 0.1f, dy = r4.y * 0.1f;
  float dw = fminf(r4.z * 0.2f, CLIPV), dh = fminf(r4.w * 0.2f, CLIPV);
  float pcx = dx * w + cx, pcy = dy * h + cy;
  float pw = __expf(dw) * w, ph = __expf(dh) * h;
  x1 = fminf(fmaxf(pcx - 0.5f * pw, 0.0f), IMGW);
  y1 = fminf(fmaxf(pcy - 0.5f * ph, 0.0f), IMGH);
  x2 = fminf(fmaxf(pcx + 0.5f * pw, 0.0f), IMGW);
  y2 = fminf(fmaxf(pcy + 0.5f * ph, 0.0f), IMGH);
}

__global__ __launch_bounds__(256) void k2_score_hist(
    const float* __restrict__ logits, const float* __restrict__ reg,
    const float* __restrict__ props, const float* __restrict__ rowmax_g,
    const float* __restrict__ inv_denom_g, float* __restrict__ scores_g,
    unsigned int* __restrict__ hist) {
  int img = blockIdx.y;
  int rem = blockIdx.x * blockDim.x + threadIdx.x;   // [0, TOTC)
  int p = rem / NCLS, c = rem % NCLS + 1;
  int row = img * NN + p;

  float sc = __expf(logits[(long)row * CCLS + c] - rowmax_g[row]) * inv_denom_g[row];
  float4 r4 = ((const float4*)(reg + (long)row * (CCLS * 4)))[c];
  float4 pr = ((const float4*)props)[row];
  float x1, y1, x2, y2;
  decode_clip(r4, pr, x1, y1, x2, y2);

  bool ok = (sc > SCORE_T) && (x2 - x1 >= MINSZ) && (y2 - y1 >= MINSZ);
  scores_g[(long)img * TOTC + rem] = ok ? sc : -1.0f;
  if (ok) {
    int b = (int)(sc * (float)NBINS);
    b = b > (NBINS - 1) ? (NBINS - 1) : (b < 0 ? 0 : b);
    atomicAdd(&hist[img * NBINS + b], 1u);
  }
}

// ---------------------------------------------------------------------------
// K3: per-image cut bin (largest bin with cumulative-from-top >= PRE).
// ---------------------------------------------------------------------------
__global__ void k3_thresh(const unsigned int* __restrict__ hist,
                          int* __restrict__ thresh_bin) {
  int img = threadIdx.x;
  if (img >= BB) return;
  int cum = 0, tb = 0;
  for (int b = NBINS - 1; b >= 0; --b) {
    cum += (int)hist[img * NBINS + b];
    if (cum >= PRE) { tb = b; break; }
  }
  thresh_bin[img] = tb;
}

// ---------------------------------------------------------------------------
// K4: compact candidates with bin >= cut bin. Grid: (TOTC/256, BB).
// ---------------------------------------------------------------------------
__global__ __launch_bounds__(256) void k4_compact(
    const float* __restrict__ scores_g, const int* __restrict__ thresh_bin,
    unsigned int* __restrict__ cnt, float* __restrict__ comp_sc,
    int* __restrict__ comp_idx) {
  int img = blockIdx.y;
  int rem = blockIdx.x * blockDim.x + threadIdx.x;
  float sc = scores_g[(long)img * TOTC + rem];
  if (sc > 0.0f) {
    int b = (int)(sc * (float)NBINS);
    b = b > (NBINS - 1) ? (NBINS - 1) : (b < 0 ? 0 : b);
    if (b >= thresh_bin[img]) {
      unsigned pos = atomicAdd(&cnt[img], 1u);
      if (pos < CAP) {
        comp_sc[img * CAP + pos]  = sc;
        comp_idx[img * CAP + pos] = rem;
      }
    }
  }
}

// ---------------------------------------------------------------------------
// K5: per-image bitonic sort (desc score, asc index tiebreak => deterministic),
// keep top PRE, re-decode exactly those boxes.
// ---------------------------------------------------------------------------
__global__ __launch_bounds__(512) void k5_sort_select(
    const unsigned int* __restrict__ cnt, const float* __restrict__ comp_sc,
    const int* __restrict__ comp_idx, const float* __restrict__ reg,
    const float* __restrict__ props, float* __restrict__ cand_sc,
    int* __restrict__ cand_lab, float* __restrict__ cand_box) {
  __shared__ float ks[CAP];
  __shared__ int   vs[CAP];
  int img = blockIdx.x, tid = threadIdx.x;
  int n = (int)cnt[img]; n = n > CAP ? CAP : n;

  for (int i = tid; i < CAP; i += 512) {
    if (i < n) { ks[i] = comp_sc[img * CAP + i]; vs[i] = comp_idx[img * CAP + i]; }
    else       { ks[i] = -1.0f;                  vs[i] = 0x7fffffff; }
  }
  __syncthreads();

  for (int k = 2; k <= CAP; k <<= 1) {
    for (int j = k >> 1; j > 0; j >>= 1) {
      for (int i = tid; i < CAP; i += 512) {
        int ixj = i ^ j;
        if (ixj > i) {
          float a = ks[i], b = ks[ixj];
          int va = vs[i], vb = vs[ixj];
          // "i comes later than ixj" in desired order (desc score, asc idx)
          bool later_i = (a < b) || (a == b && va > vb);
          bool doswap = ((i & k) == 0) ? later_i : !later_i;
          if (doswap) { ks[i] = b; ks[ixj] = a; vs[i] = vb; vs[ixj] = va; }
        }
      }
      __syncthreads();
    }
  }

  for (int i = tid; i < PRE; i += 512) {
    float sc = ks[i];
    int id = vs[i];
    float x1 = 0, y1 = 0, x2 = 0, y2 = 0;
    int lab = -1;
    if (sc > 0.0f) {
      int p = id / NCLS, c = id % NCLS + 1;
      int row = img * NN + p;
      float4 r4 = ((const float4*)(reg + (long)row * (CCLS * 4)))[c];
      float4 pr = ((const float4*)props)[row];
      decode_clip(r4, pr, x1, y1, x2, y2);
      lab = c;
    }
    int o = img * CSTRIDE + i;
    cand_sc[o] = sc; cand_lab[o] = lab;
    cand_box[o * 4 + 0] = x1; cand_box[o * 4 + 1] = y1;
    cand_box[o * 4 + 2] = x2; cand_box[o * 4 + 3] = y2;
  }
}

// ---------------------------------------------------------------------------
// K6: suppression bitmask. Label equality == batched-NMS class offsets.
// grid = BB*4 blocks of 256 threads (thread = one row i).
// ---------------------------------------------------------------------------
__global__ __launch_bounds__(256) void k6_mask(
    const int* __restrict__ cand_lab, const float* __restrict__ cand_box,
    unsigned int* __restrict__ mask) {
  __shared__ float sb[256][4];
  __shared__ int   sl[256];
  int img = blockIdx.x >> 2;
  int i = (blockIdx.x & 3) * 256 + threadIdx.x;
  bool act = i < PRE;
  float bx0 = 0, by0 = 0, bx1 = 0, by1 = 0, area_i = 0;
  int li = -1;
  if (act) {
    int o = img * CSTRIDE + i;
    bx0 = cand_box[o * 4 + 0]; by0 = cand_box[o * 4 + 1];
    bx1 = cand_box[o * 4 + 2]; by1 = cand_box[o * 4 + 3];
    li = cand_lab[o];
    area_i = (bx1 - bx0) * (by1 - by0);
  }
  unsigned bits[32];
#pragma unroll
  for (int w = 0; w < 32; ++w) bits[w] = 0u;

  for (int t0 = 0; t0 < PRE; t0 += 256) {
    int j = t0 + threadIdx.x;
    __syncthreads();
    if (j < PRE) {
      int o = img * CSTRIDE + j;
      sb[threadIdx.x][0] = cand_box[o * 4 + 0];
      sb[threadIdx.x][1] = cand_box[o * 4 + 1];
      sb[threadIdx.x][2] = cand_box[o * 4 + 2];
      sb[threadIdx.x][3] = cand_box[o * 4 + 3];
      sl[threadIdx.x] = cand_lab[o];
    } else {
      sl[threadIdx.x] = -2;
    }
    __syncthreads();
    if (act) {
      int lim = PRE - t0; lim = lim > 256 ? 256 : lim;
      for (int jj = 0; jj < lim; ++jj) {
        int jg = t0 + jj;
        if (jg <= i || sl[jj] != li) continue;
        float lx = fmaxf(bx0, sb[jj][0]), ly = fmaxf(by0, sb[jj][1]);
        float rx = fminf(bx1, sb[jj][2]), ry = fminf(by1, sb[jj][3]);
        float iw = fmaxf(rx - lx, 0.0f), ih = fmaxf(ry - ly, 0.0f);
        float inter = iw * ih;
        float area_j = (sb[jj][2] - sb[jj][0]) * (sb[jj][3] - sb[jj][1]);
        float iou = inter / fmaxf(area_i + area_j - inter, 1e-9f);
        if (iou > NMS_T) bits[jg >> 5] |= 1u << (jg & 31);
      }
    }
  }
  if (act) {
    long base = ((long)(img * PRE) + i) * 32;
#pragma unroll
    for (int w = 0; w < 32; ++w) mask[base + w] = bits[w];
  }
}

// ---------------------------------------------------------------------------
// K7: greedy scan (one wave, lane = bit-word) + finalize with stable-top_k
// filler semantics (kept in score order, then !kept in ascending index).
// ---------------------------------------------------------------------------
__global__ __launch_bounds__(128) void k7_scan_finalize(
    const float* __restrict__ cand_sc, const int* __restrict__ cand_lab,
    const float* __restrict__ cand_box, const unsigned int* __restrict__ mask,
    float* __restrict__ out) {
  __shared__ unsigned kw[32];
  __shared__ int ord[DETK];
  __shared__ int nkeep_s;
  int img = blockIdx.x, tid = threadIdx.x;

  if (tid < 32) {
    unsigned keep = 0u;
    for (int b = 0; b < 32; ++b) {
      int idx = tid * 32 + b;
      if (idx < PRE && cand_sc[img * CSTRIDE + idx] > 0.0f) keep |= 1u << b;
    }
    for (int i = 0; i < PRE; ++i) {
      unsigned kwv = __shfl(keep, i >> 5, 32);
      if (kwv & (1u << (i & 31)))
        keep &= ~mask[((long)(img * PRE) + i) * 32 + tid];
    }
    kw[tid] = keep;
  }
  __syncthreads();

  if (tid == 0) {
    int c2 = 0;
    for (int i = 0; i < PRE && c2 < DETK; ++i)
      if ((kw[i >> 5] >> (i & 31)) & 1u) ord[c2++] = i;
    nkeep_s = c2;
    for (int i = 0; i < PRE && c2 < DETK; ++i)
      if (!((kw[i >> 5] >> (i & 31)) & 1u)) ord[c2++] = i;
  }
  __syncthreads();

  if (tid < DETK) {
    int i = ord[tid];
    bool kept = tid < nkeep_s;
    int o = img * CSTRIDE + i;
    float sc = kept ? cand_sc[o] : -1.0f;
    int db = (img * DETK + tid) * 5;
    out[db + 0] = cand_box[o * 4 + 0];
    out[db + 1] = cand_box[o * 4 + 1];
    out[db + 2] = cand_box[o * 4 + 2];
    out[db + 3] = cand_box[o * 4 + 3];
    out[db + 4] = sc;
    out[BB * DETK * 5 + img * DETK + tid] = kept ? (float)cand_lab[o] : -1.0f;
    out[BB * DETK * 5 + BB * DETK + img * DETK + tid] = kept ? 1.0f : 0.0f;
  }
}

// ---------------------------------------------------------------------------
extern "C" void kernel_launch(void* const* d_in, const int* in_sizes, int n_in,
                              void* d_out, int out_size, void* d_ws, size_t ws_size,
                              hipStream_t stream) {
  (void)in_sizes; (void)n_in; (void)out_size; (void)ws_size;
  const float* logits = (const float*)d_in[0];  // [B,N,91]
  const float* reg    = (const float*)d_in[1];  // [B,N,364]
  const float* props  = (const float*)d_in[2];  // [B,N,4]
  float* out = (float*)d_out;

  char* ws = (char*)d_ws;
  size_t off = 0;
  auto carve = [&](size_t bytes) { size_t o = off; off = (off + bytes + 255) & ~(size_t)255; return o; };
  float* inv_denom  = (float*)(ws + carve((size_t)BB * NN * 4));
  float* rowmax     = (float*)(ws + carve((size_t)BB * NN * 4));
  float* scores_g   = (float*)(ws + carve((size_t)BB * TOTC * 4));
  unsigned* hist    = (unsigned*)(ws + carve((size_t)BB * NBINS * 4));
  unsigned* cnt     = (unsigned*)(ws + carve((size_t)BB * 4));
  int* thresh       = (int*)(ws + carve((size_t)BB * 4));
  float* comp_sc    = (float*)(ws + carve((size_t)BB * CAP * 4));
  int* comp_idx     = (int*)(ws + carve((size_t)BB * CAP * 4));
  float* cand_sc    = (float*)(ws + carve((size_t)BB * CSTRIDE * 4));
  int* cand_lab     = (int*)(ws + carve((size_t)BB * CSTRIDE * 4));
  float* cand_box   = (float*)(ws + carve((size_t)BB * CSTRIDE * 4 * 4));
  unsigned* nmsmask = (unsigned*)(ws + carve((size_t)BB * PRE * 32 * 4));

  hipMemsetAsync(hist, 0, (size_t)BB * NBINS * 4, stream);
  hipMemsetAsync(cnt, 0, (size_t)BB * 4, stream);

  // K1: 131072 proposals, 128 per block
  k1_softmax_stats<<<(BB * NN) / 128, 256, 0, stream>>>(logits, rowmax, inv_denom);

  dim3 cgrid(TOTC / 256, BB);   // 2880 x 16, exact cover
  k2_score_hist<<<cgrid, 256, 0, stream>>>(logits, reg, props, rowmax, inv_denom,
                                           scores_g, hist);
  k3_thresh<<<1, BB, 0, stream>>>(hist, thresh);
  k4_compact<<<cgrid, 256, 0, stream>>>(scores_g, thresh, cnt, comp_sc, comp_idx);
  k5_sort_select<<<BB, 512, 0, stream>>>(cnt, comp_sc, comp_idx, reg, props,
                                         cand_sc, cand_lab, cand_box);
  k6_mask<<<BB * 4, 256, 0, stream>>>(cand_lab, cand_box, nmsmask);
  k7_scan_finalize<<<BB, 128, 0, stream>>>(cand_sc, cand_lab, cand_box, nmsmask, out);
}
